// GGCN1_38482906972494
// MI455X (gfx1250) — matrix operand order repaired and checked by reading, not compiled, not run on hardware
//
#include <hip/hip_runtime.h>

// GGCN fused pipeline for MI455X (gfx1250, wave32, WMMA bf16 16x16x32).
// L=200000, NFEAT=J=128, S=16.
// ws layout: Hmat in bf16, L*128 elements (51.2 MB -> resident in 192MB L2,
// so the 32 gathered GEMMs of the scan never touch HBM).
// Round 2: double-buffered A-frag gathers in the scan loop so each step's
// 64-WMMA chain hides the next step's L2 gather latency.

#define LTOT 200000
#define NF 128
#define JD 128
#define SS 16
#define NTILES (LTOT / 16)          // 12500 16-row tiles
#define WPB 8                       // waves per 256-thread block

typedef __attribute__((ext_vector_type(16))) __bf16 v16bf;
typedef __attribute__((ext_vector_type(8)))  float  v8f;

union BF16x16 {
  v16bf v;
  uint4 q[2];
  __bf16 e[16];
};

// Load 16 bf16 as two 16B chunks (A-frag halves or B-frag row piece).
__device__ inline v16bf load_bf16x16(const __bf16* p0, const __bf16* p1) {
  BF16x16 u;
  u.q[0] = *reinterpret_cast<const uint4*>(p0);
  u.q[1] = *reinterpret_cast<const uint4*>(p1);
  return u.v;
}

// Convert 8 consecutive f32 -> 8 bf16.
__device__ inline void cvt8_f32_bf16(const float* p, __bf16* d) {
  float4 a = *reinterpret_cast<const float4*>(p);
  float4 b = *reinterpret_cast<const float4*>(p + 4);
  d[0] = (__bf16)a.x; d[1] = (__bf16)a.y; d[2] = (__bf16)a.z; d[3] = (__bf16)a.w;
  d[4] = (__bf16)b.x; d[5] = (__bf16)b.y; d[6] = (__bf16)b.z; d[7] = (__bf16)b.w;
}

__device__ inline v8f wmma_bf16(v16bf a, v16bf b, v8f c) {
  return __builtin_amdgcn_wmma_f32_16x16x32_bf16(false, a, false, b, (short)0, c,
                                                 false, false);
}

// Gather the 8 A-fragments (combined K=256: kc<4 -> h0 row, kc>=4 -> h1 row)
// for sample s from the L2-resident bf16 Hmat.
__device__ inline void gather_afrags(v16bf* a, const __bf16* __restrict__ H,
                                     const int* __restrict__ myIdx,
                                     int m, int hi, int s) {
  #pragma unroll
  for (int kc = 0; kc < 8; ++kc) {
    const int p = kc >> 2;
    const int r = myIdx[m * 32 + p * 16 + s];
    const __bf16* hr = H + (size_t)r * JD + (kc & 3) * 32 + hi * 8;
    a[kc] = load_bf16x16(hr, hr + 16);
  }
}

// acc[n] += relu( A(K=256) x B_combined(K=256,N=16n..) + bg ), 64 WMMAs.
__device__ inline void accum_tiles(v8f* acc, const v16bf* a,
                                   const __bf16* __restrict__ sWg,
                                   const float* bgl, int m, int hi) {
  #pragma unroll
  for (int n = 0; n < 8; ++n) {
    v8f c;
    #pragma unroll
    for (int r = 0; r < 8; ++r) c[r] = bgl[n];
    #pragma unroll
    for (int kc = 0; kc < 8; ++kc) {
      const __bf16* bp = &sWg[(n * 16 + m) * 256 + kc * 32 + hi * 8];
      c = wmma_bf16(a[kc], load_bf16x16(bp, bp + 16), c);
    }
    #pragma unroll
    for (int r = 0; r < 8; ++r) acc[n][r] += fmaxf(c[r], 0.0f);
  }
}

// ---------------------------------------------------------------------------
// Kernel 1: H = relu(X @ Wh^T + bh), stored bf16 into d_ws.
// One wave = one 16-row tile; Wh (bf16) staged in LDS (32 KB).
// ---------------------------------------------------------------------------
__global__ __launch_bounds__(256) void ggcn_hmat(
    const float* __restrict__ X, const float* __restrict__ Wh,
    const float* __restrict__ bh, __bf16* __restrict__ H) {
  __shared__ __bf16 sWh[JD * NF];
  const int tid = threadIdx.x;
  for (int i = tid; i < JD * NF; i += 256) sWh[i] = (__bf16)Wh[i];
  __syncthreads();

  const int wave = tid >> 5, lane = tid & 31;
  const int tile = blockIdx.x * WPB + wave;
  if (tile >= NTILES) return;           // wave-uniform: EXEC all-1 for WMMA
  const int m = lane & 15, hi = lane >> 4;

  // A-frags: X row -> bf16, K split into 4 chunks of 32.
  v16bf a[4];
  const float* xr = X + (size_t)(tile * 16 + m) * NF;
  #pragma unroll
  for (int kc = 0; kc < 4; ++kc) {
    BF16x16 u;
    cvt8_f32_bf16(xr + kc * 32 + hi * 8, &u.e[0]);
    cvt8_f32_bf16(xr + kc * 32 + 16 + hi * 8, &u.e[8]);
    a[kc] = u.v;
  }

  #pragma unroll
  for (int n = 0; n < 8; ++n) {
    const int jb = n * 16;
    const float bias = bh[jb + m];
    v8f c;
    #pragma unroll
    for (int r = 0; r < 8; ++r) c[r] = bias;
    #pragma unroll
    for (int kc = 0; kc < 4; ++kc) {
      const __bf16* bp = &sWh[(jb + m) * NF + kc * 32 + hi * 8];
      c = wmma_bf16(a[kc], load_bf16x16(bp, bp + 16), c);
    }
    #pragma unroll
    for (int r = 0; r < 8; ++r) {
      float v = fmaxf(c[r], 0.0f);
      H[(size_t)(tile * 16 + hi * 8 + r) * JD + jb + m] = (__bf16)v;
    }
  }
}

// ---------------------------------------------------------------------------
// Kernel 2 (fused): scan accumulation (16 steps x gathered K=256 GEMM),
// E = relu(acc/16), E2 = relu([H|E] @ [Wg0;Wg1]^T + bg), out = E2 @ Wf^T + bf.
// Dynamic LDS: combined Wg bf16 (64 KB) + per-wave idx staging (16 KB)
//            + per-wave E transpose staging (32 KB) = 112 KB.
// ---------------------------------------------------------------------------
__global__ __launch_bounds__(256) void ggcn_scan(
    const __bf16* __restrict__ H, const long long* __restrict__ pidx,
    const float* __restrict__ Wg, const float* __restrict__ bg,
    const float* __restrict__ Wf, const float* __restrict__ bfp,
    float* __restrict__ out) {
  extern __shared__ char smem[];
  __bf16* sWg = reinterpret_cast<__bf16*>(smem);                    // 128x256 bf16
  int*    sIdx = reinterpret_cast<int*>(smem + 65536);              // 8 waves * 512
  __bf16* sE  = reinterpret_cast<__bf16*>(smem + 65536 + 16384);    // 8 waves * 16x128

  const int tid = threadIdx.x;
  // Wg row-major (128 x 256): cols 0..127 = Wg0, cols 128..255 = Wg1.
  for (int i = tid; i < JD * 2 * JD; i += 256) sWg[i] = (__bf16)Wg[i];

  const int wave = tid >> 5, lane = tid & 31;
  const int tile = blockIdx.x * WPB + wave;
  int*    myIdx = sIdx + wave * 512;
  __bf16* myE   = sE + wave * 16 * JD;
  const bool active = (tile < NTILES);
  if (active) {
    const long long* pr = pidx + (size_t)tile * 16 * 2 * SS;
    for (int i = lane; i < 512; i += 32) myIdx[i] = (int)pr[i];
  }
  __syncthreads();
  if (!active) return;                  // wave-uniform exit

  const int m = lane & 15, hi = lane >> 4;

  float bgl[8];
  #pragma unroll
  for (int n = 0; n < 8; ++n) bgl[n] = bg[n * 16 + m];

  v8f acc[8];
  #pragma unroll
  for (int n = 0; n < 8; ++n) {
    #pragma unroll
    for (int r = 0; r < 8; ++r) acc[n][r] = 0.0f;
  }

  // ---- scan: acc += relu(h0@Wg0^T + h1@Wg1^T + bg), 16 samples ----
  // Software-pipelined: gathers for step s+1 issued before the WMMA chain of
  // step s, so the 64-WMMA chain hides the L2 gather latency.
  v16bf a0[8], a1[8];
  gather_afrags(a0, H, myIdx, m, hi, 0);
  #pragma unroll 1
  for (int s = 0; s < SS; s += 2) {
    gather_afrags(a1, H, myIdx, m, hi, s + 1);
    accum_tiles(acc, a0, sWg, bgl, m, hi);
    if (s + 2 < SS) gather_afrags(a0, H, myIdx, m, hi, s + 2);
    accum_tiles(acc, a1, sWg, bgl, m, hi);
  }

  // ---- E = relu(acc/16); transpose C-layout -> A-layout via LDS ----
  #pragma unroll
  for (int n = 0; n < 8; ++n) {
    #pragma unroll
    for (int r = 0; r < 8; ++r)
      myE[(r + hi * 8) * JD + n * 16 + m] = (__bf16)(acc[n][r] * 0.0625f);
  }
  asm volatile("s_wait_dscnt 0" ::: "memory");  // per-wave cross-lane LDS RAW

  // ---- final: E2 = relu([Hrow|Erow] @ [Wg0;Wg1]^T + bg); out = E2@Wf^T+bf ----
  v16bf a2[8];
  const __bf16* hrow = H + (size_t)(tile * 16 + m) * JD;
  #pragma unroll
  for (int kc = 0; kc < 4; ++kc) {
    const __bf16* p = hrow + kc * 32 + hi * 8;
    a2[kc] = load_bf16x16(p, p + 16);
  }
  #pragma unroll
  for (int kc = 0; kc < 4; ++kc) {
    const __bf16* p = myE + m * JD + kc * 32 + hi * 8;
    a2[4 + kc] = load_bf16x16(p, p + 16);
  }

  float part[8];
  #pragma unroll
  for (int r = 0; r < 8; ++r) part[r] = 0.0f;

  #pragma unroll 1
  for (int n = 0; n < 8; ++n) {
    v8f c;
    #pragma unroll
    for (int r = 0; r < 8; ++r) c[r] = bgl[n];
    #pragma unroll
    for (int kc = 0; kc < 8; ++kc) {
      const __bf16* bp = &sWg[(n * 16 + m) * 256 + kc * 32 + hi * 8];
      c = wmma_bf16(a2[kc], load_bf16x16(bp, bp + 16), c);
    }
    const float wfv = Wf[n * 16 + m];
    #pragma unroll
    for (int r = 0; r < 8; ++r) part[r] += fmaxf(c[r], 0.0f) * wfv;
  }

  // reduce over the 16 N-lanes of each half (wave32 shuffle)
  #pragma unroll
  for (int r = 0; r < 8; ++r) {
    #pragma unroll
    for (int mk = 1; mk < 16; mk <<= 1) part[r] += __shfl_xor(part[r], mk, 32);
  }
  if (m == 0) {
    const float bfv = bfp[0];
    #pragma unroll
    for (int r = 0; r < 8; ++r) out[tile * 16 + hi * 8 + r] = part[r] + bfv;
  }
}

// ---------------------------------------------------------------------------
extern "C" void kernel_launch(void* const* d_in, const int* in_sizes, int n_in,
                              void* d_out, int out_size, void* d_ws, size_t ws_size,
                              hipStream_t stream) {
  (void)in_sizes; (void)n_in; (void)out_size; (void)ws_size;
  const float*     X    = (const float*)d_in[0];
  const long long* pidx = (const long long*)d_in[1];
  const float*     Wh   = (const float*)d_in[2];
  const float*     bh   = (const float*)d_in[3];
  const float*     Wg   = (const float*)d_in[4];
  const float*     bg   = (const float*)d_in[5];
  const float*     Wf   = (const float*)d_in[6];
  const float*     bf   = (const float*)d_in[7];
  float*   out = (float*)d_out;
  __bf16*  Hm  = (__bf16*)d_ws;         // L*128 bf16 = 51.2 MB (L2-resident)

  const int blocks = (NTILES + WPB - 1) / WPB;   // 1563
  ggcn_hmat<<<blocks, 256, 0, stream>>>(X, Wh, bh, Hm);

  const size_t smem2 = 65536 + 16384 + 32768;    // 112 KB dynamic LDS
  ggcn_scan<<<blocks, 256, smem2, stream>>>(Hm, pidx, Wg, bg, Wf, bf, out);
}
